// SimpleLSTM_60713657697205
// MI455X (gfx1250) — matrix operand order repaired and checked by reading, not compiled, not run on hardware
//
#include <hip/hip_runtime.h>

// ---------------------------------------------------------------------------
// SimpleLSTM on MI455X (gfx1250): bf16 WMMA recurrent steps, fused gates.
//   N=128, L=512, HID=512, EMB=256, D=768.
//   gates = h @ W[0:512] + emb_t @ W[512:768], 4 gates fused per wave.
//
// Round-2 design: NO LDS, NO barriers in the step kernel.
//  * B (weights) pre-swizzled once into exact WMMA B-fragment layout:
//      BSwz[(((kt*128 + ct)*32 + lane)*8 + g]  (dwords)
//      lane: col = ct*16 + (lane&15); g: k = kt*32 + (g>>2)*16 + (lane>>4)*8 + 2(g&3) (+1)
//    -> per fragment: 2x global_load_b128 per lane, wave reads 1KB contig.
//  * A (h / emb, row-major bf16) loaded directly: lane l's fragment is two
//    contiguous 16B chunks of row (l&15) at k-offsets half*8 and 16+half*8.
//  * One wave per workgroup: grid (32 jTiles, 8 rowTiles) = 256 waves.
//
// Workspace layout (bytes):
//   [0,        3145728)   BSwz   bf16-pairs, 786432 dwords (3MB)
//   [3145728, 36700160)   embAll bf16 (L,N,EMB)
//   [36700160,36831232)   h ping bf16 (N,HID)
//   [36831232,36962304)   h pong bf16 (N,HID)
//   [36962304,37224448)   c state f32 (N,HID)
// ---------------------------------------------------------------------------

typedef unsigned short u16;
typedef __attribute__((ext_vector_type(16))) __bf16 v16bf;
typedef __attribute__((ext_vector_type(8)))  float  v8f;

#define LSTM_N   128
#define LSTM_L   512
#define LSTM_HID 512
#define LSTM_EMB 256
#define LSTM_D   768
#define NCOL     2048                 // 4 gates * HID
#define KTILES   (LSTM_D / 32)        // 24
#define CTILES   (NCOL / 16)          // 128
#define BSWZ_DWORDS (KTILES * CTILES * 32 * 8)   // 786432

__device__ __forceinline__ u16 f32_to_bf16(float f) {
  union { float f; unsigned u; } v; v.f = f;
  unsigned r = v.u + 0x7FFFu + ((v.u >> 16) & 1u);  // round-to-nearest-even
  return (u16)(r >> 16);
}
__device__ __forceinline__ float sigmoid_fast(float x) {
  return 1.0f / (1.0f + __expf(-x));
}
__device__ __forceinline__ float tanh_fast(float x) {
  return 1.0f - 2.0f / (__expf(2.0f * x) + 1.0f);   // exact saturation
}

union Frag {
  uint4    q[2];
  unsigned u[8];
  v16bf    v;
};

// ------------------- weight pre-swizzle into B-fragment layout -------------
// One thread per output dword.
__global__ void lstm_swz_weights(const float* __restrict__ Wi,
                                 const float* __restrict__ Wf,
                                 const float* __restrict__ Wo,
                                 const float* __restrict__ Wh,
                                 unsigned* __restrict__ BSwz) {
  int idx = blockIdx.x * blockDim.x + threadIdx.x;
  if (idx >= BSWZ_DWORDS) return;
  int g    = idx & 7;
  int lane = (idx >> 3) & 31;
  int ct   = (idx >> 8) & (CTILES - 1);
  int kt   = idx >> 15;                       // / (8*32*128)
  int col  = ct * 16 + (lane & 15);           // 0..2047
  int k    = kt * 32 + (g >> 2) * 16 + (lane >> 4) * 8 + 2 * (g & 3);
  int gate = col >> 9;
  int j    = col & 511;
  const float* W = (gate == 0) ? Wi : (gate == 1) ? Wf : (gate == 2) ? Wo : Wh;
  unsigned lo = f32_to_bf16(W[(size_t)k       * LSTM_HID + j]);
  unsigned hi = f32_to_bf16(W[(size_t)(k + 1) * LSTM_HID + j]);
  BSwz[idx] = lo | (hi << 16);
}

// --------------------------- embedding gather ------------------------------
__global__ void lstm_embed(const int* __restrict__ X,
                           const float* __restrict__ E,
                           u16* __restrict__ embAll) {
  int idx = blockIdx.x * blockDim.x + threadIdx.x;
  if (idx >= LSTM_L * LSTM_N * LSTM_EMB) return;
  int e = idx & (LSTM_EMB - 1);
  int n = (idx >> 8) & (LSTM_N - 1);
  int t = idx >> 15;
  int tok = X[n * LSTM_L + t];
  embAll[idx] = f32_to_bf16(E[(size_t)tok * LSTM_EMB + e]);
}

// ------------------------------ state init ---------------------------------
__global__ void lstm_init(u16* __restrict__ h0, float* __restrict__ c) {
  int idx = blockIdx.x * blockDim.x + threadIdx.x;
  if (idx >= LSTM_N * LSTM_HID) return;
  h0[idx] = 0;
  c[idx]  = 0.0f;
}

// ------------------------------- LSTM step ---------------------------------
// grid (32 jTiles, 8 rowTiles), block = 32 (one wave). No LDS, no barriers.
__global__ __launch_bounds__(32)
void lstm_step(const u16* __restrict__ hread,      // (N,HID) bf16 prev h
               u16* __restrict__ hwrite,           // (N,HID) bf16 new h
               const u16* __restrict__ embt,       // (N,EMB) bf16 this step
               const unsigned* __restrict__ BSwz,  // swizzled weights
               const float* __restrict__ bi, const float* __restrict__ bf_,
               const float* __restrict__ bo, const float* __restrict__ bh,
               float* __restrict__ c,              // (N,HID) f32 state
               float* __restrict__ hout) {         // (N,HID) f32 (d_out)
  const int lane = threadIdx.x;          // wave32, all lanes active
  const int half = lane >> 4;
  const int jBase   = blockIdx.x * 16;   // j within HID
  const int rowBase = blockIdx.y * 16;   // batch rows
  const int m = rowBase + (lane & 15);   // A-fragment row for this lane

  v8f acc[4];
  const v8f vzero = {0.f, 0.f, 0.f, 0.f, 0.f, 0.f, 0.f, 0.f};
  acc[0] = vzero; acc[1] = vzero; acc[2] = vzero; acc[3] = vzero;

  // Per-lane base into BSwz for this wave's column tile (gate 0).
  // gate gt adds gt*32 column-tiles = gt*32*32*8 dwords = gt*8192 dwords.
  const int ct0 = jBase >> 4;            // 0..31
  const unsigned* bptr = BSwz + ((size_t)(ct0 * 32 + lane) << 3);
  const size_t ktStride = (size_t)CTILES * 32 * 8;   // dwords per k-tile
  const size_t gStride  = (size_t)32 * 32 * 8;       // dwords per 32 col-tiles

  // ---------------- phase 1: h @ Wh (K = 0..511) ---------------------------
  {
    const u16* arow = hread + (size_t)m * LSTM_HID + half * 8;
    for (int kt = 0; kt < LSTM_HID / 32; ++kt) {
      Frag af;
      af.q[0] = *(const uint4*)(arow);
      af.q[1] = *(const uint4*)(arow + 16);
      const unsigned* bp = bptr + (size_t)kt * ktStride;
#pragma unroll
      for (int gt = 0; gt < 4; ++gt) {
        Frag bf;
        const unsigned* p = bp + (size_t)gt * gStride;
        bf.q[0] = *(const uint4*)(p);
        bf.q[1] = *(const uint4*)(p + 4);
        acc[gt] = __builtin_amdgcn_wmma_f32_16x16x32_bf16(
            false, af.v, false, bf.v, (short)0, acc[gt], false, false);
      }
      arow += 32;
    }
  }
  // ---------------- phase 2: emb_t @ Wx (K = 512..767) ---------------------
  {
    const u16* arow = embt + (size_t)m * LSTM_EMB + half * 8;
    for (int kt = 0; kt < LSTM_EMB / 32; ++kt) {
      Frag af;
      af.q[0] = *(const uint4*)(arow);
      af.q[1] = *(const uint4*)(arow + 16);
      const unsigned* bp = bptr + (size_t)(kt + LSTM_HID / 32) * ktStride;
#pragma unroll
      for (int gt = 0; gt < 4; ++gt) {
        Frag bf;
        const unsigned* p = bp + (size_t)gt * gStride;
        bf.q[0] = *(const uint4*)(p);
        bf.q[1] = *(const uint4*)(p + 4);
        acc[gt] = __builtin_amdgcn_wmma_f32_16x16x32_bf16(
            false, af.v, false, bf.v, (short)0, acc[gt], false, false);
      }
      arow += 32;
    }
  }

  // ---- fused epilogue: gates -> c,h update --------------------------------
  // C/D layout: col = lane&15, row-in-tile = half*8 + r
  const int jj = jBase + (lane & 15);
  const float Bi = bi[jj], Bf = bf_[jj], Bo = bo[jj], Bh = bh[jj];
#pragma unroll
  for (int r = 0; r < 8; ++r) {
    int n = rowBase + half * 8 + r;
    float gi = sigmoid_fast(acc[0][r] + Bi);
    float gf = sigmoid_fast(acc[1][r] + Bf);
    float go = sigmoid_fast(acc[2][r] + Bo);
    float gg = tanh_fast   (acc[3][r] + Bh);
    size_t off = (size_t)n * LSTM_HID + jj;
    float cn = gf * c[off] + gi * gg;
    float hn = go * tanh_fast(cn);
    c[off]      = cn;
    hout[off]   = hn;
    hwrite[off] = f32_to_bf16(hn);
  }
}

// ----------------------------- host launcher -------------------------------
extern "C" void kernel_launch(void* const* d_in, const int* in_sizes, int n_in,
                              void* d_out, int out_size, void* d_ws, size_t ws_size,
                              hipStream_t stream) {
  const int*   X  = (const int*)  d_in[0];
  const float* E  = (const float*)d_in[1];
  const float* Wi = (const float*)d_in[2];
  const float* bi = (const float*)d_in[3];
  const float* Wf = (const float*)d_in[4];
  const float* bf = (const float*)d_in[5];
  const float* Wo = (const float*)d_in[6];
  const float* bo = (const float*)d_in[7];
  const float* Wh = (const float*)d_in[8];
  const float* bh = (const float*)d_in[9];
  float* hout = (float*)d_out;

  char* ws = (char*)d_ws;
  unsigned* BSwz  = (unsigned*)(ws);
  u16*      embAll = (u16*)(ws + (size_t)3145728);
  u16*      hA     = (u16*)(ws + (size_t)36700160);
  u16*      hB     = (u16*)(ws + (size_t)36831232);
  float*    cState = (float*)(ws + (size_t)36962304);

  // 1) weights -> bf16, pre-swizzled B-fragment layout
  lstm_swz_weights<<<(BSWZ_DWORDS + 255) / 256, 256, 0, stream>>>(Wi, Wf, Wo, Wh, BSwz);
  // 2) embedding gather -> bf16 (L, N, EMB)
  lstm_embed<<<(LSTM_L * LSTM_N * LSTM_EMB + 255) / 256, 256, 0, stream>>>(X, E, embAll);
  // 3) zero h, c
  lstm_init<<<(LSTM_N * LSTM_HID + 255) / 256, 256, 0, stream>>>(hA, cState);

  // 4) 512 fused recurrent steps (double-buffered h), 256 single-wave WGs
  dim3 grid(LSTM_HID / 16, LSTM_N / 16);   // (32, 8)
  for (int t = 0; t < LSTM_L; ++t) {
    const u16* hr = (t & 1) ? hB : hA;
    u16*       hw = (t & 1) ? hA : hB;
    const u16* embt = embAll + (size_t)t * LSTM_N * LSTM_EMB;
    lstm_step<<<grid, 32, 0, stream>>>(hr, hw, embt, BSwz,
                                       bi, bf, bo, bh, cState, hout);
  }
}
